// CRFLayer_39444979646677
// MI455X (gfx1250) — matrix operand order repaired and checked by reading, not compiled, not run on hardware
//
#include <hip/hip_runtime.h>
#include <hip/hip_bf16.h>

// CDNA5 (gfx1250) fused dense-CRF mean-field.
// D = A(16x32 bf16, generated on the fly via exp) x B(32x16 bf16 from LDS) + C(f32)
// using v_wmma_f32_16x16x32_bf16.  Norms come free via a ones-column in q~.
// q~ chunks are staged global->LDS with GLOBAL_LOAD_ASYNC_TO_LDS_B128 (ASYNCcnt).

typedef __attribute__((ext_vector_type(16))) __bf16 v16bf;
typedef __attribute__((ext_vector_type(8)))  float  v8f;

#define CRF_N   9216
#define CRF_L   21
#define CRF_W   96
#define COLS    32          // padded label columns: 0..20 = q, 21 = ones, 22..31 = 0
#define ONES_COL 21
#define TJ      32          // j-chunk (K dim per WMMA pair)
#define NCHUNK  (CRF_N / TJ)        // 288
#define WAVES   4
#define BLOCK   (WAVES * 32)        // 128 threads
#define ROWS_WG (WAVES * 16)        // 64 rows per workgroup
#define NWG     (CRF_N / ROWS_WG)   // 144 workgroups

#define INV2G2  (-1.0f / 18.0f)     // -1/(2*3^2)
#define INV2A2  (-1.0f / 12800.0f)  // -1/(2*80^2)
#define INV2B2  (-1.0f / 338.0f)    // -1/(2*13^2)
#define W_SP    3.0f
#define W_BL    10.0f

// ---- CDNA5 async global->LDS copy (per-lane 16B), tracked by ASYNCcnt -----
__device__ __forceinline__ void async_copy_b128(const void* gsrc, void* lds_dst) {
    // Low 32 bits of a flat shared pointer == wave-relative LDS byte address
    // (ISA 10.2: LDS aperture).
    unsigned lds_addr = (unsigned)(unsigned long long)(uintptr_t)lds_dst;
    asm volatile("global_load_async_to_lds_b128 %0, %1, off"
                 :: "v"(lds_addr), "v"(gsrc) : "memory");
}
__device__ __forceinline__ void async_wait_all() {
    asm volatile("s_wait_asynccnt 0" ::: "memory");
}

// ---------------------------------------------------------------------------
// q0 = softmax(u); write transposed bf16 q~ (cols x N).  Also initialize the
// constant columns (ones + zero padding) of BOTH ping-pong buffers.
// ---------------------------------------------------------------------------
__global__ void crf_init(const float* __restrict__ u,
                         __bf16* __restrict__ qa, __bf16* __restrict__ qb) {
    int i = blockIdx.x * blockDim.x + threadIdx.x;
    if (i >= CRF_N) return;
    const float* ur = u + (size_t)i * CRF_L;
    float v[CRF_L];
    float mx = -1e30f;
#pragma unroll
    for (int l = 0; l < CRF_L; ++l) { v[l] = ur[l]; mx = fmaxf(mx, v[l]); }
    float s = 0.f;
#pragma unroll
    for (int l = 0; l < CRF_L; ++l) { v[l] = __expf(v[l] - mx); s += v[l]; }
    float inv = 1.0f / s;
#pragma unroll
    for (int l = 0; l < CRF_L; ++l)
        qa[(size_t)l * CRF_N + i] = (__bf16)(v[l] * inv);
#pragma unroll
    for (int l = CRF_L; l < COLS; ++l) {
        __bf16 c = (l == ONES_COL) ? (__bf16)1.0f : (__bf16)0.0f;
        qa[(size_t)l * CRF_N + i] = c;
        qb[(size_t)l * CRF_N + i] = c;
    }
}

// ---------------------------------------------------------------------------
// One mean-field iteration, fully fused:
//   S_sp = K_sp @ q~ , S_bl = K_bl @ q~  (column 21 = row norms)
//   q' = softmax(u + 3*S_sp/norm_sp + 10*S_bl/norm_bl)
// Each wave owns a 16-row tile; A matrices are exp()-generated per chunk.
// ---------------------------------------------------------------------------
__global__ void __launch_bounds__(BLOCK)
crf_step(const float* __restrict__ u, const float* __restrict__ rgb,
         const __bf16* __restrict__ qsrc, __bf16* __restrict__ qdst,
         float* __restrict__ outp, int write_out) {
    __shared__ __align__(32) __bf16 qch[2][COLS][TJ];     // q~ chunk, col-major
    __shared__ __align__(16) float  posch[2][TJ][2];      // (y,x) of j
    __shared__ __align__(16) float  rgbch[2][TJ][4];      // rgb of j (padded)
    __shared__ __align__(16) float  Sbuf[WAVES][2][16][COLS];

    const int tid  = threadIdx.x;
    const int wave = tid >> 5;
    const int lane = tid & 31;
    const int m    = lane & 15;          // row within wave tile / B column
    const int half = lane >> 4;          // 0: K 0..15 | 1: K 16..31
    const int baseK = half * 8;

    const int   i  = blockIdx.x * ROWS_WG + wave * 16 + m;
    const float yi = (float)(i / CRF_W);
    const float xi = (float)(i % CRF_W);
    const float cr = rgb[i * 3 + 0];
    const float cg = rgb[i * 3 + 1];
    const float cb = rgb[i * 3 + 2];

    v8f acc[4] = {};   // [0]=sp c0-15  [1]=sp c16-31  [2]=bl c0-15  [3]=bl c16-31

    auto stage = [&](int chunk, int buf) {
        const int jbase = chunk * TJ;
        // q~ chunk: 32 cols x 32 j = 1024 bf16; 128 lanes x 16B, async to LDS
        {
            const int c  = tid >> 2;
            const int jj = (tid & 3) * 8;
            const __bf16* src = qsrc + (size_t)c * CRF_N + jbase + jj;
            async_copy_b128(src, &qch[buf][c][jj]);
            // warm L2 two chunks ahead (global_prefetch_b8)
            if (jbase + 2 * TJ < CRF_N)
                __builtin_prefetch(src + 2 * TJ, 0, 1);
        }
        if (tid < TJ) {
            const int j = jbase + tid;
            posch[buf][tid][0] = (float)(j / CRF_W);
            posch[buf][tid][1] = (float)(j % CRF_W);
            rgbch[buf][tid][0] = rgb[j * 3 + 0];
            rgbch[buf][tid][1] = rgb[j * 3 + 1];
            rgbch[buf][tid][2] = rgb[j * 3 + 2];
            rgbch[buf][tid][3] = 0.f;
        }
    };

    stage(0, 0);
#pragma unroll 1
    for (int chunk = 0; chunk < NCHUNK; ++chunk) {
        const int buf = chunk & 1;
        async_wait_all();                      // own async stage of `buf` landed
        __syncthreads();                       // all waves' stages visible
        if (chunk + 1 < NCHUNK) stage(chunk + 1, buf ^ 1);

        // --- generate A matrices (16x32 bf16), contiguous K: half h holds K 16h..16h+15 ---
        v16bf a_sp, a_bl;
#pragma unroll
        for (int t = 0; t < 16; ++t) {
            const int kt = half * 16 + t;
            const float2 pj = *reinterpret_cast<const float2*>(&posch[buf][kt][0]);
            const float4 cj = *reinterpret_cast<const float4*>(&rgbch[buf][kt][0]);
            const float dy = yi - pj.x, dx = xi - pj.y;
            const float d2p = dy * dy + dx * dx;
            const float d0 = cr - cj.x, d1 = cg - cj.y, d2 = cb - cj.z;
            const float d2c = d0 * d0 + d1 * d1 + d2 * d2;
            a_sp[t] = (__bf16)__expf(d2p * INV2G2);
            a_bl[t] = (__bf16)__expf(d2p * INV2A2 + d2c * INV2B2);
        }

        // --- B tiles (32x16 bf16): 16 consecutive K values per lane/column ---
        const v16bf b0 = *reinterpret_cast<const v16bf*>(&qch[buf][m][half * 16]);
        const v16bf b1 = *reinterpret_cast<const v16bf*>(&qch[buf][16 + m][half * 16]);

        acc[0] = __builtin_amdgcn_wmma_f32_16x16x32_bf16(false, a_sp, false, b0,
                                                         (short)0, acc[0], false, false);
        acc[1] = __builtin_amdgcn_wmma_f32_16x16x32_bf16(false, a_sp, false, b1,
                                                         (short)0, acc[1], false, false);
        acc[2] = __builtin_amdgcn_wmma_f32_16x16x32_bf16(false, a_bl, false, b0,
                                                         (short)0, acc[2], false, false);
        acc[3] = __builtin_amdgcn_wmma_f32_16x16x32_bf16(false, a_bl, false, b1,
                                                         (short)0, acc[3], false, false);
    }

    // --- dump accumulators (C layout: VGPR r -> M=r (lanes 0-15) / M=8+r) ---
#pragma unroll
    for (int r = 0; r < 8; ++r) {
        const int row = half ? (8 + r) : r;
        Sbuf[wave][0][row][m]      = acc[0][r];
        Sbuf[wave][0][row][16 + m] = acc[1][r];
        Sbuf[wave][1][row][m]      = acc[2][r];
        Sbuf[wave][1][row][16 + m] = acc[3][r];
    }
    __syncthreads();

    // --- per-row update + softmax + write next q~ (and final output) ---
    if (tid < ROWS_WG) {
        const int w  = tid >> 4;
        const int mm = tid & 15;
        const int gi = blockIdx.x * ROWS_WG + tid;
        const float nsp = 1.0f / Sbuf[w][0][mm][ONES_COL];
        const float nbl = 1.0f / Sbuf[w][1][mm][ONES_COL];
        const float* ur = u + (size_t)gi * CRF_L;
        float v[CRF_L];
        float mx = -1e30f;
#pragma unroll
        for (int l = 0; l < CRF_L; ++l) {
            const float t = ur[l] + W_SP * Sbuf[w][0][mm][l] * nsp
                                  + W_BL * Sbuf[w][1][mm][l] * nbl;
            v[l] = t;
            mx = fmaxf(mx, t);
        }
        float s = 0.f;
#pragma unroll
        for (int l = 0; l < CRF_L; ++l) { v[l] = __expf(v[l] - mx); s += v[l]; }
        const float inv = 1.0f / s;
#pragma unroll
        for (int l = 0; l < CRF_L; ++l) {
            const float q = v[l] * inv;
            qdst[(size_t)l * CRF_N + gi] = (__bf16)q;
            if (write_out) outp[(size_t)gi * CRF_L + l] = q;
        }
    }
}

// ---------------------------------------------------------------------------
extern "C" void kernel_launch(void* const* d_in, const int* in_sizes, int n_in,
                              void* d_out, int out_size, void* d_ws, size_t ws_size,
                              hipStream_t stream) {
    (void)in_sizes; (void)n_in; (void)out_size; (void)ws_size;
    const float* u   = (const float*)d_in[0];   // [1, N, L] fp32
    const float* rgb = (const float*)d_in[1];   // [1, N, 3] fp32
    float* outp = (float*)d_out;                // [1, N, L] fp32

    __bf16* qa = (__bf16*)d_ws;                         // q~ ping (COLS x N bf16)
    __bf16* qb = qa + (size_t)COLS * CRF_N;             // q~ pong

    crf_init<<<(CRF_N + 255) / 256, 256, 0, stream>>>(u, qa, qb);

    __bf16* src = qa;
    __bf16* dst = qb;
    for (int it = 0; it < 5; ++it) {
        crf_step<<<NWG, BLOCK, 0, stream>>>(u, rgb, src, dst, outp,
                                            (it == 4) ? 1 : 0);
        __bf16* t = src; src = dst; dst = t;
    }
}
